// MSDeformAttnPixelDecoder_78254304133238
// MI455X (gfx1250) — matrix operand order repaired and checked by reading, not compiled, not run on hardware
//
#include <hip/hip_runtime.h>
#include <math.h>

// MSDeformAttn pixel decoder forward for gfx1250 (MI455X, wave32, WMMA).
// Fragment-direct WMMA GEMMs: A f16 [M,K] row-major, B f16 [N,K] row-major,
// per-lane fragment slices are contiguous -> global_load_b128 straight into
// WMMA operands (working set is L2-resident). Each wave register-blocks a
// 32x32 output tile (2 A-frags x 2 B-frags -> 4 independent accumulators).

typedef __attribute__((ext_vector_type(16))) _Float16 v16h;
typedef __attribute__((ext_vector_type(8)))  _Float16 v8h;
typedef __attribute__((ext_vector_type(8)))  float    v8f;
typedef __attribute__((ext_vector_type(4)))  unsigned int u32x4;
typedef __attribute__((ext_vector_type(8)))  int i32x8;
typedef __attribute__((ext_vector_type(4)))  int i32x4;

#define LTOK 5376
#define CDIM 256

#if defined(__has_builtin)
#  if __has_builtin(__builtin_amdgcn_tensor_load_to_lds)
#    define HAVE_TDM 1
#  endif
#  if __has_builtin(__builtin_amdgcn_s_wait_tensorcnt)
#    define HAVE_TWAIT 1
#  endif
#endif

// ---------------------------------------------------------------------------
// TDM warm kernel: NULL-descriptor tensor_load_to_lds (group0.count=0,
// tile_dim0=0 -> architected no-op per ISA 8.3/8.4) + s_wait_tensorcnt.
// ---------------------------------------------------------------------------
__global__ void tdm_warm_kernel()
{
#ifdef HAVE_TDM
    u32x4 g0 = {};   // count=0 => NULL tensor descriptor
    i32x8 g1 = {};
    i32x4 g2 = {};
    i32x4 g3 = {};
#  if __has_include(<hip/amd_detail/amd_gfx1250_TDM.h>)
    i32x8 g4 = {};
    __builtin_amdgcn_tensor_load_to_lds(g0, g1, g2, g3, g4, 0);   // clang-23 6-arg form
#  else
    __builtin_amdgcn_tensor_load_to_lds(g0, g1, g2, g3, 0);       // ROCm clang-22 5-arg form
#  endif
#endif
#ifdef HAVE_TWAIT
    __builtin_amdgcn_s_wait_tensorcnt(0);
#endif
}

// ---------------------------------------------------------------------------
// helpers to build fragments from two b128 loads
// ---------------------------------------------------------------------------
__device__ __forceinline__ v16h frag_a(const _Float16* p)
{
    v8h lo = *(const v8h*)p;
    v8h hi = *(const v8h*)(p + 16);
    return __builtin_shufflevector(lo, hi, 0,1,2,3,4,5,6,7,8,9,10,11,12,13,14,15);
}
__device__ __forceinline__ v16h frag_b(const _Float16* p)
{
    v8h lo = *(const v8h*)p;
    v8h hi = *(const v8h*)(p + 16);
    return __builtin_shufflevector(lo, hi, 0,1,2,3,4,5,6,7,8,9,10,11,12,13,14,15);
}

// ---------------------------------------------------------------------------
// WMMA GEMM: Out[M,N] = act(Ah @ Wh^T + bias)
// 128 threads = 4 waves (2x2); block tile 64x64; wave tile 32x32 (4 acc).
// M, N multiples of 64; K multiple of 32.
// ---------------------------------------------------------------------------
__global__ __launch_bounds__(128)
void gemm_wmma(const _Float16* __restrict__ Ah, const _Float16* __restrict__ Wh,
               const float* __restrict__ bias,
               float* __restrict__ OutF, _Float16* __restrict__ OutH,
               int M, int N, int K, int relu)
{
    const int tid  = threadIdx.x;
    const int wave = tid >> 5, lane = tid & 31;
    const int bm = blockIdx.y << 6, bn = blockIdx.x << 6;
    const int wm = (wave >> 1) << 5, wn = (wave & 1) << 5;
    const int half = lane >> 4, l15 = lane & 15;
    const _Float16* Ap0 = Ah + (size_t)(bm + wm + l15) * K + (half << 3);
    const _Float16* Ap1 = Ap0 + (size_t)16 * K;
    const _Float16* Bp0 = Wh + (size_t)(bn + wn + l15) * K + (half << 3);
    const _Float16* Bp1 = Bp0 + (size_t)16 * K;
    v8f acc00 = {}, acc01 = {}, acc10 = {}, acc11 = {};
#pragma unroll 2
    for (int k0 = 0; k0 < K; k0 += 32) {
        if (k0 + 32 < K) __builtin_prefetch(Bp0 + k0 + 32, 0, 3);  // near-scope prefetch
        v16h a0 = frag_a(Ap0 + k0);
        v16h a1 = frag_a(Ap1 + k0);
        v16h b0 = frag_b(Bp0 + k0);
        v16h b1 = frag_b(Bp1 + k0);
        acc00 = __builtin_amdgcn_wmma_f32_16x16x32_f16(false, a0, false, b0, (short)0, acc00, false, false);
        acc01 = __builtin_amdgcn_wmma_f32_16x16x32_f16(false, a0, false, b1, (short)0, acc01, false, false);
        acc10 = __builtin_amdgcn_wmma_f32_16x16x32_f16(false, a1, false, b0, (short)0, acc10, false, false);
        acc11 = __builtin_amdgcn_wmma_f32_16x16x32_f16(false, a1, false, b1, (short)0, acc11, false, false);
    }
    const int n0 = bn + wn + l15, n1 = n0 + 16;
    const float bv0 = bias ? bias[n0] : 0.0f;
    const float bv1 = bias ? bias[n1] : 0.0f;
#pragma unroll
    for (int r = 0; r < 8; ++r) {
        int m0 = bm + wm + r + (half << 3);
        int m1 = m0 + 16;
        float v00 = acc00[r] + bv0, v01 = acc01[r] + bv1;
        float v10 = acc10[r] + bv0, v11 = acc11[r] + bv1;
        if (relu) {
            v00 = fmaxf(v00, 0.0f); v01 = fmaxf(v01, 0.0f);
            v10 = fmaxf(v10, 0.0f); v11 = fmaxf(v11, 0.0f);
        }
        if (OutF) {
            OutF[(size_t)m0 * N + n0] = v00; OutF[(size_t)m0 * N + n1] = v01;
            OutF[(size_t)m1 * N + n0] = v10; OutF[(size_t)m1 * N + n1] = v11;
        }
        if (OutH) {
            OutH[(size_t)m0 * N + n0] = (_Float16)v00; OutH[(size_t)m0 * N + n1] = (_Float16)v01;
            OutH[(size_t)m1 * N + n0] = (_Float16)v10; OutH[(size_t)m1 * N + n1] = (_Float16)v11;
        }
    }
}

// ---------------------------------------------------------------------------
// conv3x3 SAME over 64x64, 256->256: 9 accumulating WMMA passes, 2x2 blocked.
// Inh f16 [4096,256]; Wt9 f16 tap-major [9][256 out][256 in]; OutF f32.
// ---------------------------------------------------------------------------
__global__ __launch_bounds__(128)
void conv3x3_wmma(const _Float16* __restrict__ Inh, const _Float16* __restrict__ Wt9,
                  float* __restrict__ OutF)
{
    const int tid  = threadIdx.x;
    const int wave = tid >> 5, lane = tid & 31;
    const int bm = blockIdx.y << 6, bn = blockIdx.x << 6;
    const int wm = (wave >> 1) << 5, wn = (wave & 1) << 5;
    const int half = lane >> 4, l15 = lane & 15;
    const int m0 = bm + wm + l15, m1 = m0 + 16;
    const int y0 = m0 >> 6, x0 = m0 & 63;
    const int y1 = m1 >> 6, x1 = m1 & 63;
    v8f acc00 = {}, acc01 = {}, acc10 = {}, acc11 = {};
    for (int tap = 0; tap < 9; ++tap) {
        const int dy = tap / 3 - 1, dx = tap % 3 - 1;
        const int ya = y0 + dy, xa = x0 + dx;
        const int yb = y1 + dy, xb = x1 + dx;
        const bool va = ((unsigned)ya < 64u) && ((unsigned)xa < 64u);
        const bool vb = ((unsigned)yb < 64u) && ((unsigned)xb < 64u);
        const _Float16* Ap0 = Inh + (size_t)((ya << 6) + xa) * 256 + (half << 3);
        const _Float16* Ap1 = Inh + (size_t)((yb << 6) + xb) * 256 + (half << 3);
        const _Float16* Bp0 = Wt9 + ((size_t)tap * 256 + (bn + wn + l15)) * 256 + (half << 3);
        const _Float16* Bp1 = Bp0 + (size_t)16 * 256;
#pragma unroll 2
        for (int k0 = 0; k0 < 256; k0 += 32) {
            v16h a0 = {}, a1 = {};
            if (va) a0 = frag_a(Ap0 + k0);
            if (vb) a1 = frag_a(Ap1 + k0);
            v16h b0 = frag_b(Bp0 + k0);
            v16h b1 = frag_b(Bp1 + k0);
            acc00 = __builtin_amdgcn_wmma_f32_16x16x32_f16(false, a0, false, b0, (short)0, acc00, false, false);
            acc01 = __builtin_amdgcn_wmma_f32_16x16x32_f16(false, a0, false, b1, (short)0, acc01, false, false);
            acc10 = __builtin_amdgcn_wmma_f32_16x16x32_f16(false, a1, false, b0, (short)0, acc10, false, false);
            acc11 = __builtin_amdgcn_wmma_f32_16x16x32_f16(false, a1, false, b1, (short)0, acc11, false, false);
        }
    }
    const int n0 = bn + wn + l15, n1 = n0 + 16;
#pragma unroll
    for (int r = 0; r < 8; ++r) {
        int mm0 = bm + wm + r + (half << 3);
        int mm1 = mm0 + 16;
        OutF[(size_t)mm0 * 256 + n0] = acc00[r];
        OutF[(size_t)mm0 * 256 + n1] = acc01[r];
        OutF[(size_t)mm1 * 256 + n0] = acc10[r];
        OutF[(size_t)mm1 * 256 + n1] = acc11[r];
    }
}

// ---------------------------------------------------------------------------
// small elementwise / conversion kernels
// ---------------------------------------------------------------------------
__global__ void cvt_f16(const float* __restrict__ s, _Float16* __restrict__ d, int n)
{
    int i = blockIdx.x * blockDim.x + threadIdx.x;
    if (i < n) d[i] = (_Float16)s[i];
}

// pad N rows (per layer) with zeros: s [layers][rows][Kk] -> d [layers][prows][Kk]
__global__ void cvt_f16_pad(const float* __restrict__ s, _Float16* __restrict__ d,
                            int rows, int prows, int Kk, int layers)
{
    int i = blockIdx.x * blockDim.x + threadIdx.x;
    if (i >= layers * prows * Kk) return;
    int l   = i / (prows * Kk);
    int rem = i - l * prows * Kk;
    int n = rem / Kk, k = rem - n * Kk;
    d[i] = (n < rows) ? (_Float16)s[((size_t)l * rows + n) * Kk + k] : (_Float16)0.0f;
}

__global__ void pad_bias(const float* __restrict__ s, float* __restrict__ d,
                         int rows, int prows, int layers)
{
    int i = blockIdx.x * blockDim.x + threadIdx.x;
    if (i >= layers * prows) return;
    int l = i / prows, n = i - l * prows;
    d[i] = (n < rows) ? s[(size_t)l * rows + n] : 0.0f;
}

__global__ void qadd_f16(const float* __restrict__ a, const float* __restrict__ b,
                         _Float16* __restrict__ d, int n)
{
    int i = blockIdx.x * blockDim.x + threadIdx.x;
    if (i < n) d[i] = (_Float16)(a[i] + b[i]);
}

// NCHW f32 -> [HW, Cin] row-major f16
__global__ void nchw_to_f16rows(const float* __restrict__ f, _Float16* __restrict__ d,
                                int HW, int Cin)
{
    int i = blockIdx.x * blockDim.x + threadIdx.x;
    if (i >= HW * Cin) return;
    int m = i / Cin, k = i - m * Cin;
    d[i] = (_Float16)f[(size_t)k * HW + m];
}

// out_w [3][O=256][I=256][3][3] f32 -> [3][tap=9][O][I] f16
__global__ void repack_conv_w(const float* __restrict__ w, _Float16* __restrict__ d)
{
    int i = blockIdx.x * blockDim.x + threadIdx.x;
    if (i >= 3 * 9 * 256 * 256) return;
    int c   = i / (9 * 65536);
    int rem = i - c * 9 * 65536;
    int tap = rem >> 16;
    int o   = (rem >> 8) & 255;
    int k   = rem & 255;
    d[i] = (_Float16)w[(size_t)c * 589824 + ((size_t)o * 256 + k) * 9 + tap];
}

// ---------------------------------------------------------------------------
// sine positional embedding, [L, 256] f32
// ---------------------------------------------------------------------------
__global__ void pos_kernel(float* __restrict__ pos)
{
    int i = blockIdx.x * blockDim.x + threadIdx.x;
    if (i >= LTOK * CDIM) return;
    int t = i >> 8, c = i & 255;
    int H, W, local;
    if (t < 4096)      { H = 64; W = 64; local = t; }
    else if (t < 5120) { H = 32; W = 32; local = t - 4096; }
    else               { H = 16; W = 16; local = t - 5120; }
    int y = local / W, x = local % W;
    const float scale = 6.283185307179586f;
    int d = c & 127;
    float base = (c < 128) ? ((float)(y + 1) / ((float)H + 1e-6f)) * scale
                           : ((float)(x + 1) / ((float)W + 1e-6f)) * scale;
    float dim_t = powf(10000.0f, 2.0f * (float)(d >> 1) / 128.0f);
    float p = base / dim_t;
    pos[i] = (d & 1) ? cosf(p) : sinf(p);
}

// ---------------------------------------------------------------------------
// softmax over 12 logits per (token, head); aw row stride = 128 (padded)
// ---------------------------------------------------------------------------
__global__ void softmax12(float* __restrict__ aw)
{
    int i = blockIdx.x * blockDim.x + threadIdx.x;
    if (i >= LTOK * 8) return;
    int t = i >> 3, h = i & 7;
    float* p = aw + (size_t)t * 128 + h * 12;
    float mx = -1e30f;
#pragma unroll
    for (int j = 0; j < 12; ++j) mx = fmaxf(mx, p[j]);
    float e[12], s = 0.0f;
#pragma unroll
    for (int j = 0; j < 12; ++j) { e[j] = expf(p[j] - mx); s += e[j]; }
    float inv = 1.0f / s;
#pragma unroll
    for (int j = 0; j < 12; ++j) p[j] = e[j] * inv;
}

// ---------------------------------------------------------------------------
// ms_deform: one wave per (token, head); lane = head dim. aw stride 128.
// ---------------------------------------------------------------------------
__global__ __launch_bounds__(128)
void msdeform_kernel(const float* __restrict__ value,
                     const float* __restrict__ off,
                     const float* __restrict__ aw,
                     _Float16* __restrict__ out)
{
    int gw   = (blockIdx.x * blockDim.x + threadIdx.x) >> 5;
    int lane = threadIdx.x & 31;
    if (gw >= LTOK * 8) return;
    int t = gw >> 3, h = gw & 7;

    int Ht, Wt, local;
    if (t < 4096)      { Ht = 64; Wt = 64; local = t; }
    else if (t < 5120) { Ht = 32; Wt = 32; local = t - 4096; }
    else               { Ht = 16; Wt = 16; local = t - 5120; }
    float ry = ((float)(local / Wt) + 0.5f) / (float)Ht;
    float rx = ((float)(local % Wt) + 0.5f) / (float)Wt;

    const int lvlW[3] = {64, 32, 16};
    const int lvlS[3] = {0, 4096, 5120};
    float acc = 0.0f;
    for (int lvl = 0; lvl < 3; ++lvl) {
        const int Wl = lvlW[lvl], Hl = lvlW[lvl], S = lvlS[lvl];
        for (int p = 0; p < 4; ++p) {
            int oi = ((h * 3 + lvl) * 4 + p) * 2;
            float ox = off[(size_t)t * 192 + oi];
            float oy = off[(size_t)t * 192 + oi + 1];
            float w  = aw[(size_t)t * 128 + (h * 3 + lvl) * 4 + p];
            float px = (rx + ox / (float)Wl) * (float)Wl - 0.5f;
            float py = (ry + oy / (float)Hl) * (float)Hl - 0.5f;
            float x0 = floorf(px), y0 = floorf(py);
            float fx = px - x0, fy = py - y0;
            int x0i = (int)x0, y0i = (int)y0;
            float s = 0.0f;
#pragma unroll
            for (int d2 = 0; d2 < 4; ++d2) {
                int dxx = d2 & 1, dyy = d2 >> 1;
                int xi = x0i + dxx, yi = y0i + dyy;
                float wgt = (dxx ? fx : 1.0f - fx) * (dyy ? fy : 1.0f - fy);
                if (xi >= 0 && xi < Wl && yi >= 0 && yi < Hl)
                    s += wgt * value[((size_t)(S + yi * Wl + xi)) * 256 + h * 32 + lane];
            }
            acc += w * s;
        }
    }
    out[(size_t)t * 256 + h * 32 + lane] = (_Float16)acc;
}

// ---------------------------------------------------------------------------
// LayerNorm over C=256 (optional residual); dual f32/f16 output.
// ---------------------------------------------------------------------------
__global__ __launch_bounds__(256)
void ln_kernel(const float* __restrict__ x, const float* __restrict__ r,
               const float* __restrict__ w, const float* __restrict__ b,
               float* __restrict__ out, _Float16* __restrict__ outH)
{
    const int t = blockIdx.x, c = threadIdx.x;
    float v = x[(size_t)t * 256 + c];
    if (r) v += r[(size_t)t * 256 + c];
    float s1 = v, s2 = v * v;
    for (int off = 16; off > 0; off >>= 1) {
        s1 += __shfl_down(s1, off);
        s2 += __shfl_down(s2, off);
    }
    __shared__ float r1[8], r2[8];
    if ((c & 31) == 0) { r1[c >> 5] = s1; r2[c >> 5] = s2; }
    __syncthreads();
    __shared__ float mS, iS;
    if (c == 0) {
        float t1 = 0.0f, t2 = 0.0f;
        for (int k = 0; k < 8; ++k) { t1 += r1[k]; t2 += r2[k]; }
        float m   = t1 / 256.0f;
        float var = t2 / 256.0f - m * m;
        mS = m; iS = rsqrtf(var + 1e-5f);
    }
    __syncthreads();
    float o = (v - mS) * iS * w[c] + b[c];
    out[(size_t)t * 256 + c] = o;
    if (outH) outH[(size_t)t * 256 + c] = (_Float16)o;
}

// ---------------------------------------------------------------------------
// GroupNorm(32) + ReLU over [4096, 256]: f32 in place + f16 copy.
// ---------------------------------------------------------------------------
__global__ __launch_bounds__(256)
void gnorm_relu(float* __restrict__ buf, const float* __restrict__ w,
                const float* __restrict__ b, _Float16* __restrict__ outH)
{
    const int g = blockIdx.x, tid = threadIdx.x;
    float s1 = 0.0f, s2 = 0.0f;
    for (int i = tid; i < 4096 * 8; i += 256) {
        int tok = i >> 3, cc = (g << 3) + (i & 7);
        float v = buf[(size_t)tok * 256 + cc];
        s1 += v; s2 += v * v;
    }
    for (int off = 16; off > 0; off >>= 1) {
        s1 += __shfl_down(s1, off);
        s2 += __shfl_down(s2, off);
    }
    __shared__ float r1[8], r2[8];
    if ((tid & 31) == 0) { r1[tid >> 5] = s1; r2[tid >> 5] = s2; }
    __syncthreads();
    __shared__ float mS, iS;
    if (tid == 0) {
        float t1 = 0.0f, t2 = 0.0f;
        for (int k = 0; k < 8; ++k) { t1 += r1[k]; t2 += r2[k]; }
        float m   = t1 / 32768.0f;
        float var = t2 / 32768.0f - m * m;
        mS = m; iS = rsqrtf(var + 1e-5f);
    }
    __syncthreads();
    for (int i = tid; i < 4096 * 8; i += 256) {
        int tok = i >> 3, cc = (g << 3) + (i & 7);
        size_t idx = (size_t)tok * 256 + cc;
        float v = (buf[idx] - mS) * iS * w[cc] + b[cc];
        v = fmaxf(v, 0.0f);
        buf[idx]  = v;
        outH[idx] = (_Float16)v;
    }
}

// ---------------------------------------------------------------------------
// [HW, C] token-major f32 -> NCHW (C, HW)
// ---------------------------------------------------------------------------
__global__ void tok2chw(const float* __restrict__ rows, float* __restrict__ dst, int HW)
{
    int i = blockIdx.x * blockDim.x + threadIdx.x;
    if (i >= HW * 256) return;
    int c = i / HW, hw = i - c * HW;
    dst[i] = rows[(size_t)hw * 256 + c];
}

// ---------------------------------------------------------------------------
// Host orchestration
// ---------------------------------------------------------------------------
static inline void launch_gemm(const _Float16* A, const _Float16* W, const float* bias,
                               float* OutF, _Float16* OutH, int M, int N, int K,
                               int relu, hipStream_t s)
{
    dim3 g((unsigned)(N / 64), (unsigned)(M / 64));
    gemm_wmma<<<g, dim3(128), 0, s>>>(A, W, bias, OutF, OutH, M, N, K, relu);
}

extern "C" void kernel_launch(void* const* d_in, const int* in_sizes, int n_in,
                              void* d_out, int out_size, void* d_ws, size_t ws_size,
                              hipStream_t stream)
{
    (void)in_sizes; (void)n_in; (void)out_size; (void)ws_size;
    const float* feat0 = (const float*)d_in[0];
    const float* feat1 = (const float*)d_in[1];
    const float* feat2 = (const float*)d_in[2];
    const float* in0_w = (const float*)d_in[3];
    const float* in0_b = (const float*)d_in[4];
    const float* in1_w = (const float*)d_in[5];
    const float* in1_b = (const float*)d_in[6];
    const float* in2_w = (const float*)d_in[7];
    const float* in2_b = (const float*)d_in[8];
    const float* off_w = (const float*)d_in[9];
    const float* off_b = (const float*)d_in[10];
    const float* aw_w  = (const float*)d_in[11];
    const float* aw_b  = (const float*)d_in[12];
    const float* vp_w  = (const float*)d_in[13];
    const float* vp_b  = (const float*)d_in[14];
    const float* op_w  = (const float*)d_in[15];
    const float* op_b  = (const float*)d_in[16];
    const float* n1_w  = (const float*)d_in[17];
    const float* n1_b  = (const float*)d_in[18];
    const float* ff1_w = (const float*)d_in[19];
    const float* ff1_b = (const float*)d_in[20];
    const float* ff2_w = (const float*)d_in[21];
    const float* ff2_b = (const float*)d_in[22];
    const float* n2_w  = (const float*)d_in[23];
    const float* n2_b  = (const float*)d_in[24];
    const float* fn_w  = (const float*)d_in[25];
    const float* fn_b  = (const float*)d_in[26];
    const float* lat_w = (const float*)d_in[27];
    const float* lat_b = (const float*)d_in[28];
    const float* out_w = (const float*)d_in[29];
    const float* gnw   = (const float*)d_in[30];
    const float* gnb   = (const float*)d_in[31];
    float* outp = (float*)d_out;

    // ---- workspace: f32 region then f16 region ----
    float* ws = (float*)d_ws;
    size_t of = 0;
    float* pos   = ws + of; of += (size_t)LTOK * CDIM;
    float* src   = ws + of; of += (size_t)LTOK * CDIM;
    float* offb  = ws + of; of += (size_t)LTOK * 192;
    float* awb   = ws + of; of += (size_t)LTOK * 128;   // padded N=128
    float* val   = ws + of; of += (size_t)LTOK * CDIM;
    float* resid = ws + of; of += (size_t)LTOK * CDIM;
    float* cbuf  = ws + of; of += (size_t)4096 * CDIM;
    float* awbp  = ws + of; of += (size_t)6 * 128;      // padded aw bias
    float* srcF  = val;                 // val is dead after the encoder loop

    _Float16* hb = (_Float16*)(ws + of);
    size_t oh = 0;
    _Float16* featT0 = hb + oh; oh += (size_t)4096 * 256;
    _Float16* featT1 = hb + oh; oh += (size_t)1024 * 512;
    _Float16* featT2 = hb + oh; oh += (size_t)256 * 1024;
    _Float16* qh     = hb + oh; oh += (size_t)LTOK * 256;
    _Float16* srch   = hb + oh; oh += (size_t)LTOK * 256;
    _Float16* samph  = hb + oh; oh += (size_t)LTOK * 256;
    _Float16* ffhh   = hb + oh; oh += (size_t)LTOK * 1024;
    _Float16* mfh    = hb + oh; oh += (size_t)4096 * 256;
    _Float16* in0h   = hb + oh; oh += (size_t)256 * 256;
    _Float16* in1h   = hb + oh; oh += (size_t)256 * 512;
    _Float16* in2h   = hb + oh; oh += (size_t)256 * 1024;
    _Float16* offh   = hb + oh; oh += (size_t)6 * 192 * 256;
    _Float16* awh    = hb + oh; oh += (size_t)6 * 128 * 256;  // padded N=128
    _Float16* vph    = hb + oh; oh += (size_t)6 * 256 * 256;
    _Float16* oph    = hb + oh; oh += (size_t)6 * 256 * 256;
    _Float16* ff1h   = hb + oh; oh += (size_t)6 * 1024 * 256;
    _Float16* ff2h   = hb + oh; oh += (size_t)6 * 256 * 1024;
    _Float16* lath   = hb + oh; oh += (size_t)256 * 256;
    _Float16* cw9h   = hb + oh; oh += (size_t)3 * 9 * 256 * 256;
    _Float16* srcFh  = qh;              // qh is dead after the encoder loop

    // TDM warm (NULL descriptor no-op, exercises tensor path)
    tdm_warm_kernel<<<dim3(1), dim3(32), 0, stream>>>();

    // ---- one-time f16 conversions ----
    auto cvtN = [&](const float* s, _Float16* d, int n) {
        cvt_f16<<<dim3((n + 255) / 256), dim3(256), 0, stream>>>(s, d, n);
    };
    cvtN(in0_w, in0h, 256 * 256);
    cvtN(in1_w, in1h, 256 * 512);
    cvtN(in2_w, in2h, 256 * 1024);
    cvtN(off_w, offh, 6 * 192 * 256);
    cvt_f16_pad<<<dim3((6 * 128 * 256 + 255) / 256), dim3(256), 0, stream>>>(
        aw_w, awh, 96, 128, 256, 6);
    pad_bias<<<dim3((6 * 128 + 255) / 256), dim3(256), 0, stream>>>(aw_b, awbp, 96, 128, 6);
    cvtN(vp_w,  vph,  6 * 256 * 256);
    cvtN(op_w,  oph,  6 * 256 * 256);
    cvtN(ff1_w, ff1h, 6 * 1024 * 256);
    cvtN(ff2_w, ff2h, 6 * 256 * 1024);
    cvtN(lat_w, lath, 256 * 256);
    repack_conv_w<<<dim3((3 * 9 * 65536 + 255) / 256), dim3(256), 0, stream>>>(out_w, cw9h);
    nchw_to_f16rows<<<dim3((4096 * 256 + 255) / 256), dim3(256), 0, stream>>>(feat0, featT0, 4096, 256);
    nchw_to_f16rows<<<dim3((1024 * 512 + 255) / 256), dim3(256), 0, stream>>>(feat1, featT1, 1024, 512);
    nchw_to_f16rows<<<dim3((256 * 1024 + 255) / 256), dim3(256), 0, stream>>>(feat2, featT2, 256, 1024);

    pos_kernel<<<dim3((LTOK * CDIM + 255) / 256), dim3(256), 0, stream>>>(pos);

    // ---- input projections -> src (f32) + srch (f16) ----
    launch_gemm(featT0, in0h, in0_b, src,              srch,              4096, 256, 256,  0, stream);
    launch_gemm(featT1, in1h, in1_b, src + 4096 * 256, srch + 4096 * 256, 1024, 256, 512,  0, stream);
    launch_gemm(featT2, in2h, in2_b, src + 5120 * 256, srch + 5120 * 256,  256, 256, 1024, 0, stream);

    // ---- 6 encoder layers ----
    for (int i = 0; i < 6; ++i) {
        qadd_f16<<<dim3(LTOK), dim3(256), 0, stream>>>(src, pos, qh, LTOK * 256);
        launch_gemm(qh,   offh + (size_t)i * 192 * 256, off_b + (size_t)i * 192, offb, nullptr, LTOK, 192, 256, 0, stream);
        launch_gemm(qh,   awh  + (size_t)i * 128 * 256, awbp  + (size_t)i * 128, awb,  nullptr, LTOK, 128, 256, 0, stream);
        launch_gemm(srch, vph  + (size_t)i * 65536,     vp_b  + (size_t)i * 256, val,  nullptr, LTOK, 256, 256, 0, stream);
        softmax12<<<dim3((LTOK * 8 + 255) / 256), dim3(256), 0, stream>>>(awb);
        msdeform_kernel<<<dim3(LTOK * 8 * 32 / 128), dim3(128), 0, stream>>>(val, offb, awb, samph);
        launch_gemm(samph, oph + (size_t)i * 65536, op_b + (size_t)i * 256, resid, nullptr, LTOK, 256, 256, 0, stream);
        ln_kernel<<<dim3(LTOK), dim3(256), 0, stream>>>(src, resid,
            n1_w + (size_t)i * 256, n1_b + (size_t)i * 256, src, srch);
        launch_gemm(srch, ff1h + (size_t)i * 262144, ff1_b + (size_t)i * 1024, nullptr, ffhh, LTOK, 1024, 256, 1, stream);
        launch_gemm(ffhh, ff2h + (size_t)i * 262144, ff2_b + (size_t)i * 256,  resid,  nullptr, LTOK, 256, 1024, 0, stream);
        ln_kernel<<<dim3(LTOK), dim3(256), 0, stream>>>(src, resid,
            n2_w + (size_t)i * 256, n2_b + (size_t)i * 256, src, srch);
    }

    // ---- final LN -> srcF (f32) + srcFh (f16) ----
    ln_kernel<<<dim3(LTOK), dim3(256), 0, stream>>>(src, nullptr, fn_w, fn_b, srcF, srcFh);

    // mems -> d_out (NCHW). Output order: mf, mems0, mems1, mems2
    tok2chw<<<dim3((4096 * 256 + 255) / 256), dim3(256), 0, stream>>>(srcF,              outp + 1048576, 4096);
    tok2chw<<<dim3((1024 * 256 + 255) / 256), dim3(256), 0, stream>>>(srcF + 4096 * 256, outp + 2097152, 1024);
    tok2chw<<<dim3(( 256 * 256 + 255) / 256), dim3(256), 0, stream>>>(srcF + 5120 * 256, outp + 2359296,  256);

    // lateral 1x1 conv on mems0 (f16 out only; feeds conv chain)
    launch_gemm(srcFh, lath, lat_b, nullptr, mfh, 4096, 256, 256, 0, stream);

    // 3x (conv3x3 -> GroupNorm(32) -> ReLU)
    for (int i = 0; i < 3; ++i) {
        conv3x3_wmma<<<dim3(4, 64), dim3(128), 0, stream>>>(
            mfh, cw9h + (size_t)i * 589824, cbuf);
        gnorm_relu<<<dim3(32), dim3(256), 0, stream>>>(
            cbuf, gnw + (size_t)i * 256, gnb + (size_t)i * 256, mfh);
    }
    tok2chw<<<dim3((4096 * 256 + 255) / 256), dim3(256), 0, stream>>>(cbuf, outp, 4096);
}
